// Pndb_43344809951805
// MI455X (gfx1250) — matrix operand order repaired and checked by reading, not compiled, not run on hardware
//
#include <hip/hip_runtime.h>
#include <math.h>

#define B_N 16
#define S_N 2048
#define D_N 1024
#define Q_N 64
#define BS_N (B_N * S_N)   // 32768

typedef _Float16 h16;
typedef __attribute__((ext_vector_type(16))) _Float16 v16h;
typedef __attribute__((ext_vector_type(8)))  _Float16 v8h;
typedef __attribute__((ext_vector_type(4)))  _Float16 v4h;
typedef __attribute__((ext_vector_type(8)))  float    v8f;
typedef __attribute__((ext_vector_type(4)))  int      vi4;

#if defined(__has_builtin)
#  if __has_builtin(__builtin_amdgcn_global_load_async_to_lds_b128)
#    define HAVE_ASYNC_LDS 1
#  endif
#  if __has_builtin(__builtin_amdgcn_s_wait_asynccnt)
#    define HAVE_WAIT_ASYNC 1
#  endif
#endif

static __device__ inline v8f vzero8() {
  v8f z = {0.f,0.f,0.f,0.f,0.f,0.f,0.f,0.f};
  return z;
}
static __device__ inline float sigmoidf_(float x) { return 1.f / (1.f + expf(-x)); }

// 16B global -> LDS copy; async (ASYNCcnt) when the toolchain exposes it.
// Builtin prototype (from compiler diagnostic): arg0 = AS(1) int4* (global src),
// arg1 = AS(3) int4* (LDS dst), then imm offset, imm cpol.
static __device__ inline void copy16_g2l(h16* lds, const h16* g) {
#ifdef HAVE_ASYNC_LDS
  __attribute__((address_space(1))) vi4* gp =
      (__attribute__((address_space(1))) vi4*)const_cast<h16*>(g);
  __attribute__((address_space(3))) vi4* lp =
      (__attribute__((address_space(3))) vi4*)lds;
  __builtin_amdgcn_global_load_async_to_lds_b128(gp, lp, 0, 0);
#else
  *(uint4*)lds = *(const uint4*)g;
#endif
}
static __device__ inline void wait_async_copies() {
#ifdef HAVE_ASYNC_LDS
#  ifdef HAVE_WAIT_ASYNC
  __builtin_amdgcn_s_wait_asynccnt(0);
#  else
  asm volatile("s_wait_asynccnt 0x0" ::: "memory");
#  endif
#endif
}

static __device__ inline v16h concat8(v8h lo, v8h hi) {
  return __builtin_shufflevector(lo, hi, 0,1,2,3,4,5,6,7,8,9,10,11,12,13,14,15);
}
// A-frag from row-major tile (ld multiple of 8): 16 contiguous K at fixed m.
static __device__ inline v16h a_frag(const h16* base, int ld, int mr, int half) {
  const h16* rp = base + mr * ld;
  v8h lo = *(const v8h*)(rp + 16 * half);
  v8h hi = *(const v8h*)(rp + 16 * half + 8);
  return concat8(lo, hi);
}
// B-frag from [n][k] tile (ldk multiple of 8): 16 contiguous K at fixed n.
static __device__ inline v16h b_frag_col(const h16* base, int ldk, int n, int koff, int half) {
  const h16* p = base + n * ldk + koff + 16 * half;
  v8h lo = *(const v8h*)(p);
  v8h hi = *(const v8h*)(p + 8);
  return concat8(lo, hi);
}
static __device__ inline v8f wmma_f16(v16h a, v16h b, v8f c) {
  return __builtin_amdgcn_wmma_f32_16x16x32_f16(false, a, false, b, (short)0, c, false, false);
}
static __device__ inline void store4h(h16* p, float4 v) {
  v4h t = {(h16)v.x, (h16)v.y, (h16)v.z, (h16)v.w};
  *(v4h*)p = t;
}

// ---- K1: Qk = questions@Wk, Qo = questions@Wo  (64x1024, f32 out) ----
__global__ void k_proj(const float* __restrict__ qs, const float* __restrict__ Wk,
                       const float* __restrict__ Wo, float* __restrict__ Qkf,
                       float* __restrict__ Qof) {
  const float* W  = blockIdx.y ? Wo : Wk;
  float*      Out = blockIdx.y ? Qof : Qkf;
  int nBase = blockIdx.x * 64;
  __shared__ __align__(16) h16 As[Q_N][40];   // A tile row-major [m=q][k]
  __shared__ __align__(16) h16 Bn[64][40];    // B tile [n][k]
  int tid = threadIdx.x, wave = tid >> 5, lane = tid & 31, half = lane >> 4, mr = lane & 15;
  v8f acc[4] = {vzero8(), vzero8(), vzero8(), vzero8()};
  for (int kk = 0; kk < D_N; kk += 32) {
#pragma unroll
    for (int it = 0; it < 4; ++it) {              // questions chunk 64x32 -> row-major
      int s = tid + it * 128;
      int row = s >> 3, c4 = (s & 7) * 4;
      float4 v = *(const float4*)(qs + row * D_N + kk + c4);
      store4h(&As[row][c4], v);
    }
#pragma unroll
    for (int it = 0; it < 4; ++it) {              // W chunk 32k x 64n -> [n][k]
      int s = tid + it * 128;
      int r = s >> 4, c4 = (s & 15) * 4;
      float4 v = *(const float4*)(W + (size_t)(kk + r) * D_N + nBase + c4);
      Bn[c4+0][r]=(h16)v.x; Bn[c4+1][r]=(h16)v.y; Bn[c4+2][r]=(h16)v.z; Bn[c4+3][r]=(h16)v.w;
    }
    __syncthreads();
    v16h a = a_frag(&As[wave*16][0], 40, mr, half);
    v16h bfr[4];
#pragma unroll
    for (int nt = 0; nt < 4; ++nt) bfr[nt] = b_frag_col(&Bn[0][0], 40, nt*16 + mr, 0, half);
#pragma unroll
    for (int nt = 0; nt < 4; ++nt) acc[nt] = wmma_f16(a, bfr[nt], acc[nt]);
    __syncthreads();
  }
#pragma unroll
  for (int nt = 0; nt < 4; ++nt)
#pragma unroll
    for (int r = 0; r < 8; ++r) {
      int m = wave*16 + half*8 + r;
      Out[m * D_N + nBase + nt*16 + mr] = acc[nt][r];
    }
}

// ---- K2: convert Qk/Qo to f16 ----
__global__ void k_qconv(const float* __restrict__ Qkf, const float* __restrict__ Qof,
                        h16* __restrict__ Qkh, h16* __restrict__ Qoh) {
  int i = blockIdx.x * 256 + threadIdx.x;
  Qkh[i] = (h16)Qkf[i];
  Qoh[i] = (h16)Qof[i];
}

// ---- K3: qbk[q] = questions[q]·bk, qbo[q] = questions[q]·bo ----
__global__ void k_qbias(const float* __restrict__ qs, const float* __restrict__ bk,
                        const float* __restrict__ bo, float* __restrict__ qbk,
                        float* __restrict__ qbo) {
  const float* bias = blockIdx.y ? bo : bk;
  float*       out  = blockIdx.y ? qbo : qbk;
  int wave = threadIdx.x >> 5, lane = threadIdx.x & 31;
  int q = blockIdx.x * 4 + wave;
  float s = 0.f;
  for (int i = lane; i < D_N; i += 32) s += qs[q * D_N + i] * bias[i];
#pragma unroll
  for (int off = 16; off; off >>= 1) s += __shfl_xor(s, off, 32);
  if (lane == 0) out[q] = s;
}

// ---- K4: phase-1 scores Sc[q][bs] = (raw[bs]·Qk[q] + qbk[q])/sqrt(D), mask; gi[bs] ----
__global__ void k_p1_scores(const float* __restrict__ raw, const unsigned char* __restrict__ mask,
                            const h16* __restrict__ Qkh, const float* __restrict__ qbk,
                            const float* __restrict__ Wi, const float* __restrict__ bi,
                            float* __restrict__ Sc, float* __restrict__ gi) {
  int nBase = blockIdx.x * 64;                       // bs tile
  __shared__ __align__(16) h16 Qs[Q_N][40];          // A tile: Qk rows
  __shared__ __align__(16) h16 Bn[64][40];           // B tile [n=bs][k=d] (natural raw rows)
  int tid = threadIdx.x, wave = tid >> 5, lane = tid & 31, half = lane >> 4, mr = lane & 15;
  int r0 = tid >> 3, c4 = (tid & 7) * 4;             // fixed loader slots
  float gacc[4] = {0.f, 0.f, 0.f, 0.f};
  v8f acc[4] = {vzero8(), vzero8(), vzero8(), vzero8()};
  for (int d0 = 0; d0 < D_N; d0 += 32) {
#pragma unroll
    for (int it = 0; it < 2; ++it) {                 // Qk f16 chunk 64x32, async b128 copies
      int s = tid + it * 128;
      int row = s >> 2, c8 = (s & 3) * 8;
      copy16_g2l(&Qs[row][c8], Qkh + row * D_N + d0 + c8);
    }
    float4 w = *(const float4*)(Wi + d0 + c4);
#pragma unroll
    for (int rr = 0; rr < 4; ++rr) {                 // raw chunk 64x32 + gi partial
      int row = r0 + rr * 16;
      float4 v = *(const float4*)(raw + (size_t)(nBase + row) * D_N + d0 + c4);
      gacc[rr] += v.x*w.x + v.y*w.y + v.z*w.z + v.w*w.w;
      store4h(&Bn[row][c4], v);                      // natural orientation, packed b64 store
    }
    if (d0 + 32 < D_N)
      __builtin_prefetch(raw + (size_t)(nBase + r0) * D_N + d0 + 32 + c4, 0, 1);
    wait_async_copies();
    __syncthreads();
    v16h a = a_frag(&Qs[wave*16][0], 40, mr, half);
    v16h bfr[4];
#pragma unroll
    for (int nt = 0; nt < 4; ++nt) bfr[nt] = b_frag_col(&Bn[0][0], 40, nt*16 + mr, 0, half);
#pragma unroll
    for (int nt = 0; nt < 4; ++nt) acc[nt] = wmma_f16(a, bfr[nt], acc[nt]);
    __syncthreads();
  }
  float bi0 = bi[0];
#pragma unroll
  for (int rr = 0; rr < 4; ++rr) {                   // shuffle-reduce 8 lanes/row
    float s = gacc[rr];
    s += __shfl_xor(s, 1, 32); s += __shfl_xor(s, 2, 32); s += __shfl_xor(s, 4, 32);
    if ((tid & 7) == 0) gi[nBase + r0 + rr*16] = sigmoidf_(s + bi0);
  }
  const float inv_sqrt_d = 0.03125f;                 // 1/sqrt(1024)
#pragma unroll
  for (int nt = 0; nt < 4; ++nt)
#pragma unroll
    for (int r = 0; r < 8; ++r) {
      int q = wave*16 + half*8 + r;
      int bs = nBase + nt*16 + mr;
      float sc = (acc[nt][r] + qbk[q]) * inv_sqrt_d;
      if (!mask[bs]) sc = -1e9f;
      Sc[(size_t)q * BS_N + bs] = sc;
    }
}

// ---- K5: softmax over s per (b,q), multiply by gi, in place ----
__global__ void k_p1_softmax(float* __restrict__ Sc, const float* __restrict__ gi) {
  int q = blockIdx.x & 63, b = blockIdx.x >> 6;
  float* row = Sc + (size_t)q * BS_N + b * S_N;
  const float* g = gi + b * S_N;
  int tid = threadIdx.x;
  float x[8], mx = -3.4e38f;
#pragma unroll
  for (int i = 0; i < 8; ++i) { x[i] = row[tid + i*256]; mx = fmaxf(mx, x[i]); }
  __shared__ float red[256];
  red[tid] = mx; __syncthreads();
  for (int off = 128; off; off >>= 1) { if (tid < off) red[tid] = fmaxf(red[tid], red[tid+off]); __syncthreads(); }
  mx = red[0]; __syncthreads();
  float sm = 0.f;
#pragma unroll
  for (int i = 0; i < 8; ++i) { x[i] = expf(x[i] - mx); sm += x[i]; }
  red[tid] = sm; __syncthreads();
  for (int off = 128; off; off >>= 1) { if (tid < off) red[tid] += red[tid+off]; __syncthreads(); }
  float inv = 1.f / red[0];
#pragma unroll
  for (int i = 0; i < 8; ++i) row[tid + i*256] = x[i] * inv * g[tid + i*256];
}

// ---- K6: A partials: Apart[chunk] = W[:,chunk] @ raw[chunk,:] ----
__global__ void k_p1_accum(const float* __restrict__ raw, const float* __restrict__ Sc,
                           float* __restrict__ Apart) {
  int nBase = blockIdx.x * 64;                       // d tile
  int kBase = blockIdx.y * 1024;                     // bs chunk
  __shared__ __align__(16) h16 Ws[Q_N][40];          // A tile: weight rows [q][k]
  __shared__ __align__(16) h16 Bn[64][40];           // B tile [n=d][k=bs] (transposed store)
  int tid = threadIdx.x, wave = tid >> 5, lane = tid & 31, half = lane >> 4, mr = lane & 15;
  v8f acc[4] = {vzero8(), vzero8(), vzero8(), vzero8()};
  for (int k0 = 0; k0 < 1024; k0 += 32) {
    int kb = kBase + k0;
#pragma unroll
    for (int it = 0; it < 4; ++it) {                 // weights chunk 64x32 (f32->f16)
      int s = tid + it * 128;
      int row = s >> 3, c4 = (s & 7) * 4;
      float4 v = *(const float4*)(Sc + (size_t)row * BS_N + kb + c4);
      store4h(&Ws[row][c4], v);
    }
#pragma unroll
    for (int it = 0; it < 4; ++it) {                 // raw chunk 32k x 64n -> [n][k]
      int s = tid + it * 128;
      int r = s >> 4, c4 = (s & 15) * 4;
      float4 v = *(const float4*)(raw + (size_t)(kb + r) * D_N + nBase + c4);
      Bn[c4+0][r]=(h16)v.x; Bn[c4+1][r]=(h16)v.y; Bn[c4+2][r]=(h16)v.z; Bn[c4+3][r]=(h16)v.w;
    }
    if (k0 + 32 < 1024)
      __builtin_prefetch(raw + (size_t)(kb + 32 + (tid >> 4)) * D_N + nBase + (tid & 15) * 4, 0, 1);
    __syncthreads();
    v16h a = a_frag(&Ws[wave*16][0], 40, mr, half);
    v16h bfr[4];
#pragma unroll
    for (int nt = 0; nt < 4; ++nt) bfr[nt] = b_frag_col(&Bn[0][0], 40, nt*16 + mr, 0, half);
#pragma unroll
    for (int nt = 0; nt < 4; ++nt) acc[nt] = wmma_f16(a, bfr[nt], acc[nt]);
    __syncthreads();
  }
  float* dst = Apart + (size_t)blockIdx.y * (Q_N * D_N);
#pragma unroll
  for (int nt = 0; nt < 4; ++nt)
#pragma unroll
    for (int r = 0; r < 8; ++r) {
      int q = wave*16 + half*8 + r;
      dst[q * D_N + nBase + nt*16 + mr] = acc[nt][r];
    }
}

// ---- K7: reduce partials -> A (with 1/B), deterministic ----
__global__ void k_areduce(const float* __restrict__ Apart, float* __restrict__ Af) {
  int i = blockIdx.x * 256 + threadIdx.x;            // 65536 elements
  float s = 0.f;
#pragma unroll 4
  for (int c = 0; c < 32; ++c) s += Apart[c * (Q_N * D_N) + i];
  Af[i] = s * (1.f / (float)B_N);
}

// ---- K8: A -> f16 transposed copy AhT[d][q] + aw = A @ Wu2 ----
__global__ void k_aconv(const float* __restrict__ Af, const float* __restrict__ Wu2,
                        h16* __restrict__ AhT, float* __restrict__ aw) {
  int wave = threadIdx.x >> 5, lane = threadIdx.x & 31;
  int q = blockIdx.x * 4 + wave;
  float s = 0.f;
  for (int i = lane; i < D_N; i += 32) {
    float a = Af[q * D_N + i];
    AhT[(size_t)i * Q_N + q] = (h16)a;               // transposed for k_p2_out B frags
    s += a * Wu2[i];
  }
#pragma unroll
  for (int off = 16; off; off >>= 1) s += __shfl_xor(s, off, 32);
  if (lane == 0) aw[q] = s;
}

// ---- K9: phase-2 scores + softmax over q -> P[bs][q] f16; g1[bs] ----
__global__ void k_p2_scores(const float* __restrict__ pd, const h16* __restrict__ Qoh,
                            const float* __restrict__ qbo, const float* __restrict__ Wu1,
                            const float* __restrict__ bu1, float* __restrict__ g1,
                            h16* __restrict__ P) {
  int nBase = blockIdx.x * 64;
  __shared__ __align__(16) h16 Qs[Q_N][40];
  __shared__ __align__(16) h16 Bn[64][40];
  __shared__ float Sf[Q_N][65];
  __shared__ float mxs[64], sinv[64];
  int tid = threadIdx.x, wave = tid >> 5, lane = tid & 31, half = lane >> 4, mr = lane & 15;
  int r0 = tid >> 3, c4 = (tid & 7) * 4;
  float gacc[4] = {0.f, 0.f, 0.f, 0.f};
  v8f acc[4] = {vzero8(), vzero8(), vzero8(), vzero8()};
  for (int d0 = 0; d0 < D_N; d0 += 32) {
#pragma unroll
    for (int it = 0; it < 2; ++it) {
      int s = tid + it * 128;
      int row = s >> 2, c8 = (s & 3) * 8;
      copy16_g2l(&Qs[row][c8], Qoh + row * D_N + d0 + c8);
    }
    float4 w = *(const float4*)(Wu1 + d0 + c4);
#pragma unroll
    for (int rr = 0; rr < 4; ++rr) {
      int row = r0 + rr * 16;
      float4 v = *(const float4*)(pd + (size_t)(nBase + row) * D_N + d0 + c4);
      gacc[rr] += v.x*w.x + v.y*w.y + v.z*w.z + v.w*w.w;
      store4h(&Bn[row][c4], v);
    }
    if (d0 + 32 < D_N)
      __builtin_prefetch(pd + (size_t)(nBase + r0) * D_N + d0 + 32 + c4, 0, 1);
    wait_async_copies();
    __syncthreads();
    v16h a = a_frag(&Qs[wave*16][0], 40, mr, half);
    v16h bfr[4];
#pragma unroll
    for (int nt = 0; nt < 4; ++nt) bfr[nt] = b_frag_col(&Bn[0][0], 40, nt*16 + mr, 0, half);
#pragma unroll
    for (int nt = 0; nt < 4; ++nt) acc[nt] = wmma_f16(a, bfr[nt], acc[nt]);
    __syncthreads();
  }
  float bu10 = bu1[0];
#pragma unroll
  for (int rr = 0; rr < 4; ++rr) {
    float s = gacc[rr];
    s += __shfl_xor(s, 1, 32); s += __shfl_xor(s, 2, 32); s += __shfl_xor(s, 4, 32);
    if ((tid & 7) == 0) g1[nBase + r0 + rr*16] = s + bu10;
  }
  const float inv_sqrt_q = 0.125f;                   // 1/sqrt(64)
#pragma unroll
  for (int nt = 0; nt < 4; ++nt)
#pragma unroll
    for (int r = 0; r < 8; ++r) {
      int q = wave*16 + half*8 + r;
      Sf[q][nt*16 + mr] = (acc[nt][r] + qbo[q]) * inv_sqrt_q;
    }
  __syncthreads();
  if (tid < 64) {                                    // softmax over q per bs column
    float m = -3.4e38f;
    for (int q = 0; q < Q_N; ++q) m = fmaxf(m, Sf[q][tid]);
    float s = 0.f;
    for (int q = 0; q < Q_N; ++q) s += expf(Sf[q][tid] - m);
    mxs[tid] = m; sinv[tid] = 1.f / s;
  }
  __syncthreads();
#pragma unroll
  for (int it = 0; it < 32; ++it) {                  // write P[bs][q], coalesced over q
    int s = tid + it * 128;
    int row = s >> 6, q = s & 63;
    P[(size_t)(nBase + row) * Q_N + q] = (h16)(expf(Sf[q][row] - mxs[row]) * sinv[row]);
  }
}

// ---- K10: out = post_dec + (P @ A) * sigmoid(g1 + P·aw + bu2 + b1) ----
__global__ void k_p2_out(const float* __restrict__ pd, const h16* __restrict__ P,
                         const h16* __restrict__ AhT, const float* __restrict__ aw,
                         const float* __restrict__ g1, const float* __restrict__ bu2,
                         const float* __restrict__ b1, float* __restrict__ out) {
  int mBase = blockIdx.x * 16;                       // 16 bs rows per block
  __shared__ __align__(16) h16 Ps[16][72];
  __shared__ float gate[16];
  int tid = threadIdx.x, wave = tid >> 5, lane = tid & 31, half = lane >> 4, mr = lane & 15;
  {
    int row = tid >> 3, c8 = (tid & 7) * 8;          // P tile 16x64 (async b128)
    copy16_g2l(&Ps[row][c8], P + (size_t)(mBase + row) * Q_N + c8);
  }
  wait_async_copies();
  __syncthreads();
  if (tid < 16) {
    float g2 = 0.f;
    for (int q = 0; q < Q_N; ++q) g2 += (float)Ps[tid][q] * aw[q];
    gate[tid] = sigmoidf_(g1[mBase + tid] + g2 + bu2[0] + b1[0]);
  }
  __syncthreads();
  v16h a0 = a_frag(&Ps[0][0], 72, mr, half);         // K = 0..31
  v16h a1 = a_frag(&Ps[0][32], 72, mr, half);        // K = 32..63
  int nWave = wave * 256;
  for (int nt = 0; nt < 16; ++nt) {
    int nG = nWave + nt * 16;
    // B frags straight from global AhT[d][q]: 16 contiguous q at fixed d (b128 loads)
    const h16* bp = AhT + (size_t)(nG + mr) * Q_N;
    v16h b0 = concat8(*(const v8h*)(bp + 16*half),      *(const v8h*)(bp + 16*half + 8));
    v16h b1f = concat8(*(const v8h*)(bp + 32 + 16*half), *(const v8h*)(bp + 32 + 16*half + 8));
    v8f acc = vzero8();
    acc = wmma_f16(a0, b0, acc);
    acc = wmma_f16(a1, b1f, acc);
#pragma unroll
    for (int r = 0; r < 8; ++r) {
      int m = half * 8 + r;
      size_t idx = (size_t)(mBase + m) * D_N + nG + mr;
      out[idx] = pd[idx] + acc[r] * gate[m];
    }
  }
}

extern "C" void kernel_launch(void* const* d_in, const int* in_sizes, int n_in,
                              void* d_out, int out_size, void* d_ws, size_t ws_size,
                              hipStream_t stream) {
  (void)in_sizes; (void)n_in; (void)out_size; (void)ws_size;
  const float* raw        = (const float*)d_in[0];
  const float* post_dec   = (const float*)d_in[1];
  const unsigned char* mk = (const unsigned char*)d_in[2];
  const float* questions  = (const float*)d_in[3];
  const float* Wk  = (const float*)d_in[4];
  const float* bk  = (const float*)d_in[5];
  const float* Wi  = (const float*)d_in[6];
  const float* bi  = (const float*)d_in[7];
  const float* Wo  = (const float*)d_in[8];
  const float* bo  = (const float*)d_in[9];
  const float* Wu1 = (const float*)d_in[10];
  const float* bu1 = (const float*)d_in[11];
  const float* Wu2 = (const float*)d_in[12];
  const float* bu2 = (const float*)d_in[13];
  const float* b1  = (const float*)d_in[14];
  float* out = (float*)d_out;

  char* ws = (char*)d_ws;
  size_t off = 0;
  auto alloc = [&](size_t bytes) -> void* {
    void* p = ws + off;
    off = (off + bytes + 255) & ~(size_t)255;
    return p;
  };
  float* Qkf   = (float*)alloc((size_t)Q_N * D_N * 4);
  float* Qof   = (float*)alloc((size_t)Q_N * D_N * 4);
  float* Af    = (float*)alloc((size_t)Q_N * D_N * 4);
  h16*   Qkh   = (h16*)  alloc((size_t)Q_N * D_N * 2);
  h16*   Qoh   = (h16*)  alloc((size_t)Q_N * D_N * 2);
  h16*   AhT   = (h16*)  alloc((size_t)Q_N * D_N * 2);
  float* qbk   = (float*)alloc(Q_N * 4);
  float* qbo   = (float*)alloc(Q_N * 4);
  float* aw    = (float*)alloc(Q_N * 4);
  float* gi    = (float*)alloc((size_t)BS_N * 4);
  float* g1    = (float*)alloc((size_t)BS_N * 4);
  float* Sc    = (float*)alloc((size_t)Q_N * BS_N * 4);     // 8 MB
  float* Apart = (float*)alloc((size_t)32 * Q_N * D_N * 4); // 8 MB
  h16*   P     = (h16*)  alloc((size_t)BS_N * Q_N * 2);     // 4 MB

  k_proj      <<<dim3(16, 2),  128, 0, stream>>>(questions, Wk, Wo, Qkf, Qof);
  k_qconv     <<<256,          256, 0, stream>>>(Qkf, Qof, Qkh, Qoh);
  k_qbias     <<<dim3(16, 2),  128, 0, stream>>>(questions, bk, bo, qbk, qbo);
  k_p1_scores <<<BS_N / 64,    128, 0, stream>>>(raw, mk, Qkh, qbk, Wi, bi, Sc, gi);
  k_p1_softmax<<<B_N * Q_N,    256, 0, stream>>>(Sc, gi);
  k_p1_accum  <<<dim3(16, 32), 128, 0, stream>>>(raw, Sc, Apart);
  k_areduce   <<<256,          256, 0, stream>>>(Apart, Af);
  k_aconv     <<<16,           128, 0, stream>>>(Af, Wu2, AhT, aw);
  k_p2_scores <<<BS_N / 64,    128, 0, stream>>>(post_dec, Qoh, qbo, Wu1, bu1, g1, P);
  k_p2_out    <<<BS_N / 16,    128, 0, stream>>>(post_dec, P, AhT, aw, g1, bu2, b1, out);
}
